// OverfitJobGNN_60662118088919
// MI455X (gfx1250) — matrix operand order repaired and checked by reading, not compiled, not run on hardware
//
#include <hip/hip_runtime.h>
#include <hip/hip_bf16.h>

// ---------------------------------------------------------------------------
// GNN forward for MI455X (gfx1250, wave32).
//  * Dense GEMMs (x@W) run on v_wmma_f32_16x16x32_bf16: f32->bf16 convert on
//    load, f32 accumulate. K=128 hardcoded -> fully unrolled (4 WMMA / tile).
//  * Weight panel B (K x 16 per block) is staged once into LDS, pre-swizzled
//    into the WMMA B lane layout, so the K-loop reads it with ds_load_b128.
//  * A fragments load as float4 (global_load_b128) + v_cvt_pk_bf16_f32.
//  * Edge aggregation (GCN norm-sum, GAT softmax) is scatter-add via f32
//    global atomics; node-feature matrices (25-51MB) are L2-resident (192MB).
//  * GAT segment-max uses an order-preserving float<->uint encode + atomicMax.
// ---------------------------------------------------------------------------

typedef __attribute__((ext_vector_type(16))) __bf16 v16bf;
typedef __attribute__((ext_vector_type(8)))  float  v8f;

#define HDIM  128
#define NHEAD 2
#define OUTC  32
#define KK    128      // contraction dim of every GEMM in this net

// ------------------------------ small helpers ------------------------------
__device__ __forceinline__ float leaky02(float x) { return x > 0.0f ? x : 0.2f * x; }
// order-preserving float -> uint encode (for atomicMax over signed floats)
__device__ __forceinline__ unsigned fenc(float f) {
  unsigned u = __float_as_uint(f);
  return (u & 0x80000000u) ? ~u : (u | 0x80000000u);
}
__device__ __forceinline__ float fdec(unsigned u) {
  return (u & 0x80000000u) ? __uint_as_float(u & 0x7fffffffu) : __uint_as_float(~u);
}

__global__ void fill_f32_kernel(float* p, float v, int n) {
  int i = blockIdx.x * blockDim.x + threadIdx.x;
  if (i < n) p[i] = v;
}
__global__ void fill_u32_kernel(unsigned* p, unsigned v, int n) {
  int i = blockIdx.x * blockDim.x + threadIdx.x;
  if (i < n) p[i] = v;
}

// ------------------------------ degree / norm ------------------------------
__global__ void deg_kernel(const int* __restrict__ dst, float* __restrict__ deg, int E) {
  int e = blockIdx.x * blockDim.x + threadIdx.x;
  if (e < E) atomicAdd(&deg[dst[e]], 1.0f);
}
__global__ void rsqrt_kernel(float* d, int n) {
  int i = blockIdx.x * blockDim.x + threadIdx.x;
  if (i < n) d[i] = rsqrtf(d[i]);   // deg >= 1 always (self loop)
}

// ------------------------------ WMMA GEMM ----------------------------------
// C[M,NW] = (RELU ? relu(A) : A) @ W[KK,NW]   (bf16 multiply, f32 accumulate)
// Block = 256 threads = 8 waves; block covers 128 rows x 16 cols of C.
// All 8 waves share one 16-column B panel staged in LDS in WMMA lane layout.
template <int NW, int RELU>
__global__ void __launch_bounds__(256)
gemm_wmma_kernel(const float* __restrict__ A, const float* __restrict__ W,
                 float* __restrict__ C, int M)
{
  constexpr int NTILES = NW / 16;
  const int bm = blockIdx.x / NTILES;        // 128-row block index
  const int tn = blockIdx.x % NTILES;        // 16-col tile index

  // --- stage B panel (KK x 16) into LDS, swizzled to WMMA B lane layout:
  //     lds_b[ks][lane][i] = B[ks*32 + 8*(lane>>4) + (i&7) + 16*(i>>3)][lane&15]
  __shared__ __attribute__((aligned(32))) __bf16 lds_b[KK / 32][32][16];
  {
    const int k = threadIdx.x;               // threads 0..127 handle one K-row each
    if (k < KK) {
      const float* Wrow = W + (size_t)k * NW + tn * 16;
      const int ks = k >> 5;
      const int h  = (k >> 3) & 1;           // lane half
      const int i  = (k & 7) | ((k >> 1) & 8); // position inside 32B fragment
#pragma unroll
      for (int c = 0; c < 16; ++c)
        lds_b[ks][h * 16 + c][i] = (__bf16)Wrow[c];
    }
  }
  __syncthreads();

  const int wave = threadIdx.x >> 5;
  const int lane = threadIdx.x & 31;
  const int tm   = bm * 8 + wave;            // 16-row tile index
  if (tm * 16 >= M) return;                  // wave-uniform exit (EXEC all-1s)

  const int row  = lane & 15;
  const int half = lane >> 4;
  int m_in = tm * 16 + row; if (m_in >= M) m_in = M - 1;
  const float* Ap = A + (size_t)m_in * KK;
  const int col = tn * 16 + (lane & 15);
  const int ka  = half * 8;                  // A: lanes16-31 start at K=+8

  v8f acc = {};
#pragma unroll
  for (int ks = 0; ks < KK / 32; ++ks) {
    const int kk = ks * 32;
    // A fragment: two 8-float runs -> four global_load_b128
    float4 a0 = *(const float4*)(Ap + kk + ka);
    float4 a1 = *(const float4*)(Ap + kk + ka + 4);
    float4 a2 = *(const float4*)(Ap + kk + ka + 16);
    float4 a3 = *(const float4*)(Ap + kk + ka + 20);
    v16bf a;
    if (RELU) {
      a[0]=(__bf16)fmaxf(a0.x,0.f); a[1]=(__bf16)fmaxf(a0.y,0.f);
      a[2]=(__bf16)fmaxf(a0.z,0.f); a[3]=(__bf16)fmaxf(a0.w,0.f);
      a[4]=(__bf16)fmaxf(a1.x,0.f); a[5]=(__bf16)fmaxf(a1.y,0.f);
      a[6]=(__bf16)fmaxf(a1.z,0.f); a[7]=(__bf16)fmaxf(a1.w,0.f);
      a[8]=(__bf16)fmaxf(a2.x,0.f); a[9]=(__bf16)fmaxf(a2.y,0.f);
      a[10]=(__bf16)fmaxf(a2.z,0.f); a[11]=(__bf16)fmaxf(a2.w,0.f);
      a[12]=(__bf16)fmaxf(a3.x,0.f); a[13]=(__bf16)fmaxf(a3.y,0.f);
      a[14]=(__bf16)fmaxf(a3.z,0.f); a[15]=(__bf16)fmaxf(a3.w,0.f);
    } else {
      a[0]=(__bf16)a0.x; a[1]=(__bf16)a0.y; a[2]=(__bf16)a0.z; a[3]=(__bf16)a0.w;
      a[4]=(__bf16)a1.x; a[5]=(__bf16)a1.y; a[6]=(__bf16)a1.z; a[7]=(__bf16)a1.w;
      a[8]=(__bf16)a2.x; a[9]=(__bf16)a2.y; a[10]=(__bf16)a2.z; a[11]=(__bf16)a2.w;
      a[12]=(__bf16)a3.x; a[13]=(__bf16)a3.y; a[14]=(__bf16)a3.z; a[15]=(__bf16)a3.w;
    }
    // B fragment: contiguous 32B per lane -> two ds_load_b128
    v16bf b = *(const v16bf*)(&lds_b[ks][lane][0]);
    acc = __builtin_amdgcn_wmma_f32_16x16x32_bf16(
        /*neg_a=*/false, a, /*neg_b=*/false, b,
        /*c_mod=*/(short)0, acc, /*reuse_a=*/false, /*reuse_b=*/false);
  }

  // C/D layout: VGPR j -> row tm*16 + half*8 + j, this lane's column
  float* Cp = C + (size_t)(tm * 16 + half * 8) * NW + col;
  if (tm * 16 + 16 <= M) {                   // uniform fast path, no guards
#pragma unroll
    for (int j = 0; j < 8; ++j) Cp[(size_t)j * NW] = acc[j];
  } else {
#pragma unroll
    for (int j = 0; j < 8; ++j)
      if (tm * 16 + half * 8 + j < M) Cp[(size_t)j * NW] = acc[j];
  }
}

// ------------------------------ GCN aggregation ----------------------------
// out[i,:] = b + dinv[i]^2 * h[i,:]        (self-loop term + bias as init)
__global__ void gcn_selfinit_kernel(const float* __restrict__ h, const float* __restrict__ bias,
                                    const float* __restrict__ dinv, float* __restrict__ outp, int n)
{
  long long i = (long long)blockIdx.x * blockDim.x + threadIdx.x;
  if (i >= (long long)n * HDIM) return;
  int node = (int)(i >> 7);
  int c    = (int)(i & (HDIM - 1));
  float di = dinv[node];
  outp[i] = bias[c] + di * di * h[i];
}

// out[dst,:] += dinv[src]*dinv[dst] * h[src,:]   (one wave per edge, float4/lane)
__global__ void gcn_edge_kernel(const float* __restrict__ h, const int* __restrict__ src,
                                const int* __restrict__ dst, const float* __restrict__ dinv,
                                float* __restrict__ outp, int E)
{
  long long tid = (long long)blockIdx.x * blockDim.x + threadIdx.x;
  long long e = tid >> 5;
  if (e >= E) return;
  int l = (int)(tid & 31);
  int s = src[e], d = dst[e];
  float norm = dinv[s] * dinv[d];
  float4 v = ((const float4*)(h + (size_t)s * HDIM))[l];
  float* o = outp + (size_t)d * HDIM + l * 4;
  atomicAdd(o + 0, norm * v.x);
  atomicAdd(o + 1, norm * v.y);
  atomicAdd(o + 2, norm * v.z);
  atomicAdd(o + 3, norm * v.w);
}

// ------------------------------ GAT ----------------------------------------
__global__ void gat_att_kernel(const float* __restrict__ hg, const float* __restrict__ attS,
                               const float* __restrict__ attD, float* __restrict__ alS,
                               float* __restrict__ alD, int n)
{
  int i = blockIdx.x * blockDim.x + threadIdx.x;     // node*NHEAD
  if (i >= n * NHEAD) return;
  int node = i >> 1, hd = i & 1;
  const float* f  = hg + (size_t)node * (NHEAD * HDIM) + hd * HDIM;
  const float* as = attS + hd * HDIM;
  const float* ad = attD + hd * HDIM;
  float s = 0.0f, d = 0.0f;
  for (int k = 0; k < HDIM; ++k) { float v = f[k]; s += v * as[k]; d += v * ad[k]; }
  alS[i] = s; alD[i] = d;
}

// pass 1: per-dst segment max of leaky_relu(alS[src]+alD[dst]); edges [0,E+n)
__global__ void gat_max_kernel(const int* __restrict__ src, const int* __restrict__ dst,
                               const float* __restrict__ alS, const float* __restrict__ alD,
                               unsigned* __restrict__ mMax, int E, int n)
{
  long long i = (long long)blockIdx.x * blockDim.x + threadIdx.x;
  if (i >= (long long)(E + n) * NHEAD) return;
  long long e = i >> 1; int hd = (int)(i & 1);
  int s = (e < E) ? src[e] : (int)(e - E);
  int d = (e < E) ? dst[e] : (int)(e - E);
  float a = leaky02(alS[s * NHEAD + hd] + alD[d * NHEAD + hd]);
  atomicMax(&mMax[d * NHEAD + hd], fenc(a));
}

// pass 2: denom[dst] += exp(alpha - m[dst])
__global__ void gat_denom_kernel(const int* __restrict__ src, const int* __restrict__ dst,
                                 const float* __restrict__ alS, const float* __restrict__ alD,
                                 const unsigned* __restrict__ mMax, float* __restrict__ denom,
                                 int E, int n)
{
  long long i = (long long)blockIdx.x * blockDim.x + threadIdx.x;
  if (i >= (long long)(E + n) * NHEAD) return;
  long long e = i >> 1; int hd = (int)(i & 1);
  int s = (e < E) ? src[e] : (int)(e - E);
  int d = (e < E) ? dst[e] : (int)(e - E);
  float a = leaky02(alS[s * NHEAD + hd] + alD[d * NHEAD + hd]);
  atomicAdd(&denom[d * NHEAD + hd], __expf(a - fdec(mMax[d * NHEAD + hd])));
}

// pass 3: acc[dst,:] += sum_heads 0.5 * coef_h * hg[src, h*128 + :]
//         (head-mean folded into the 0.5 weight)
__global__ void gat_accum_kernel(const int* __restrict__ src, const int* __restrict__ dst,
                                 const float* __restrict__ alS, const float* __restrict__ alD,
                                 const unsigned* __restrict__ mMax, const float* __restrict__ denom,
                                 const float* __restrict__ hg, float* __restrict__ acc,
                                 int E, int n)
{
  long long tid = (long long)blockIdx.x * blockDim.x + threadIdx.x;
  long long e = tid >> 5;
  if (e >= (long long)E + n) return;
  int l = (int)(tid & 31);
  int s = (e < E) ? src[e] : (int)(e - E);
  int d = (e < E) ? dst[e] : (int)(e - E);
  float a0 = leaky02(alS[s * NHEAD + 0] + alD[d * NHEAD + 0]);
  float a1 = leaky02(alS[s * NHEAD + 1] + alD[d * NHEAD + 1]);
  float w0 = 0.5f * __expf(a0 - fdec(mMax[d * NHEAD + 0])) / (denom[d * NHEAD + 0] + 1e-16f);
  float w1 = 0.5f * __expf(a1 - fdec(mMax[d * NHEAD + 1])) / (denom[d * NHEAD + 1] + 1e-16f);
  const float4* f0 = (const float4*)(hg + (size_t)s * (NHEAD * HDIM));
  const float4* f1 = f0 + (HDIM / 4);
  float4 v0 = f0[l], v1 = f1[l];
  float* o = acc + (size_t)d * HDIM + l * 4;
  atomicAdd(o + 0, w0 * v0.x + w1 * v1.x);
  atomicAdd(o + 1, w0 * v0.y + w1 * v1.y);
  atomicAdd(o + 2, w0 * v0.z + w1 * v1.z);
  atomicAdd(o + 3, w0 * v0.w + w1 * v1.w);
}

__global__ void gat_fin_kernel(float* __restrict__ a, const float* __restrict__ bias, int n)
{
  long long i = (long long)blockIdx.x * blockDim.x + threadIdx.x;
  if (i >= (long long)n * HDIM) return;
  int c = (int)(i & (HDIM - 1));
  a[i] = fmaxf(a[i] + bias[c], 0.0f);
}

// ------------------------------ pooling ------------------------------------
__global__ void pool_kernel(const float* __restrict__ h, const int* __restrict__ batch,
                            float* __restrict__ sums, float* __restrict__ cnts, int n)
{
  long long tid = (long long)blockIdx.x * blockDim.x + threadIdx.x;
  long long node = tid >> 5;
  if (node >= n) return;
  int l = (int)(tid & 31);
  int b = batch[node];
  float4 v = ((const float4*)(h + (size_t)node * HDIM))[l];
  float* o = sums + (size_t)b * HDIM + l * 4;
  atomicAdd(o + 0, v.x); atomicAdd(o + 1, v.y);
  atomicAdd(o + 2, v.z); atomicAdd(o + 3, v.w);
  if (l == 0) atomicAdd(&cnts[b], 1.0f);
}

// ------------------------------ MLP head + log_softmax ---------------------
__global__ void head_kernel(const float* __restrict__ sums, const float* __restrict__ cnts,
                            const float* __restrict__ fc1W, const float* __restrict__ fc1b,
                            const float* __restrict__ fc2W, const float* __restrict__ fc2b,
                            float* __restrict__ out)
{
  __shared__ float grow[HDIM];
  __shared__ float hrow[HDIM];
  __shared__ float logits[OUTC];
  __shared__ float red[2];
  int g = blockIdx.x, t = threadIdx.x;
  float scale = 1.0f / fmaxf(cnts[g], 1.0f) + 1.0f;   // mean-pool + add-pool share sums
  grow[t] = sums[g * HDIM + t] * scale;
  __syncthreads();
  float a = fc1b[t];
  for (int k = 0; k < HDIM; ++k) a += grow[k] * fc1W[k * HDIM + t];
  hrow[t] = fmaxf(a, 0.0f);
  __syncthreads();
  if (t < OUTC) {
    float z = fc2b[t];
    for (int k = 0; k < HDIM; ++k) z += hrow[k] * fc2W[k * OUTC + t];
    logits[t] = z;
  }
  __syncthreads();
  if (t == 0) {
    float m = -INFINITY;
    for (int o = 0; o < OUTC; ++o) m = fmaxf(m, logits[o]);
    float ssum = 0.0f;
    for (int o = 0; o < OUTC; ++o) ssum += __expf(logits[o] - m);
    red[0] = m; red[1] = logf(ssum);
  }
  __syncthreads();
  if (t < OUTC) out[g * OUTC + t] = logits[t] - red[0] - red[1];
}

// ------------------------------ host orchestration -------------------------
extern "C" void kernel_launch(void* const* d_in, const int* in_sizes, int n_in,
                              void* d_out, int out_size, void* d_ws, size_t ws_size,
                              hipStream_t stream)
{
  const float* x     = (const float*)d_in[0];
  const int*   ei    = (const int*)d_in[1];
  const int*   batch = (const int*)d_in[2];
  const float* W1 = (const float*)d_in[3];  const float* b1 = (const float*)d_in[4];
  const float* W2 = (const float*)d_in[5];  const float* b2 = (const float*)d_in[6];
  const float* W3 = (const float*)d_in[7];  const float* b3 = (const float*)d_in[8];
  const float* gatW = (const float*)d_in[9];
  const float* attS = (const float*)d_in[10];
  const float* attD = (const float*)d_in[11];
  const float* gatB = (const float*)d_in[12];
  const float* fc1W = (const float*)d_in[13]; const float* fc1b = (const float*)d_in[14];
  const float* fc2W = (const float*)d_in[15]; const float* fc2b = (const float*)d_in[16];
  float* out = (float*)d_out;
  (void)n_in; (void)ws_size;

  const int NN = in_sizes[2];          // nodes
  const int E  = in_sizes[1] / 2;      // edges
  const int G  = out_size / OUTC;      // graphs
  const int* src = ei;
  const int* dst = ei + E;

  // workspace carve-out (256B aligned): ~104 MB total
  char* ws = (char*)d_ws; size_t off = 0;
  auto salloc = [&](size_t bytes) -> void* {
    void* p = ws + off; off += (bytes + 255) & ~(size_t)255; return p;
  };
  float*    bufA  = (float*)salloc((size_t)NN * HDIM * 4);           // gemm out / GAT acc
  float*    bufB  = (float*)salloc((size_t)NN * HDIM * 4);           // agg out / gemm in
  float*    hGat  = (float*)salloc((size_t)NN * NHEAD * HDIM * 4);   // GAT features
  float*    dinv  = (float*)salloc((size_t)NN * 4);
  float*    alS   = (float*)salloc((size_t)NN * NHEAD * 4);
  float*    alD   = (float*)salloc((size_t)NN * NHEAD * 4);
  unsigned* mMax  = (unsigned*)salloc((size_t)NN * NHEAD * 4);
  float*    denom = (float*)salloc((size_t)NN * NHEAD * 4);
  float*    sums  = (float*)salloc((size_t)G * HDIM * 4);
  float*    cnts  = (float*)salloc((size_t)G * 4);

  auto blk = [](long long t) { return (unsigned)((t + 255) / 256); };

  // ---- symmetric normalization: deg includes self-loop -> init 1.0
  fill_f32_kernel<<<blk(NN), 256, 0, stream>>>(dinv, 1.0f, NN);
  deg_kernel<<<blk(E), 256, 0, stream>>>(dst, dinv, E);
  rsqrt_kernel<<<blk(NN), 256, 0, stream>>>(dinv, NN);

  const unsigned gemmBlk  = (unsigned)(((NN + 127) / 128) * (HDIM / 16));
  const unsigned gatBlk   = (unsigned)(((NN + 127) / 128) * (NHEAD * HDIM / 16));

  // ---- GCN layer 1 (no relu on input x)
  gemm_wmma_kernel<HDIM, 0><<<gemmBlk, 256, 0, stream>>>(x, W1, bufA, NN);
  gcn_selfinit_kernel<<<blk((long long)NN * HDIM), 256, 0, stream>>>(bufA, b1, dinv, bufB, NN);
  gcn_edge_kernel<<<blk((long long)E * 32), 256, 0, stream>>>(bufA, src, dst, dinv, bufB, E);
  // ---- GCN layer 2 (relu fused into gemm A-read)
  gemm_wmma_kernel<HDIM, 1><<<gemmBlk, 256, 0, stream>>>(bufB, W2, bufA, NN);
  gcn_selfinit_kernel<<<blk((long long)NN * HDIM), 256, 0, stream>>>(bufA, b2, dinv, bufB, NN);
  gcn_edge_kernel<<<blk((long long)E * 32), 256, 0, stream>>>(bufA, src, dst, dinv, bufB, E);
  // ---- GCN layer 3
  gemm_wmma_kernel<HDIM, 1><<<gemmBlk, 256, 0, stream>>>(bufB, W3, bufA, NN);
  gcn_selfinit_kernel<<<blk((long long)NN * HDIM), 256, 0, stream>>>(bufA, b3, dinv, bufB, NN);
  gcn_edge_kernel<<<blk((long long)E * 32), 256, 0, stream>>>(bufA, src, dst, dinv, bufB, E);

  // ---- GAT: transform (relu'd) features to 2 heads x 128
  gemm_wmma_kernel<NHEAD * HDIM, 1><<<gatBlk, 256, 0, stream>>>(bufB, gatW, hGat, NN);
  gat_att_kernel<<<blk((long long)NN * NHEAD), 256, 0, stream>>>(hGat, attS, attD, alS, alD, NN);
  fill_u32_kernel<<<blk((long long)NN * NHEAD), 256, 0, stream>>>(mMax, 0x007FFFFFu, NN * NHEAD); // enc(-inf)
  gat_max_kernel<<<blk((long long)(E + NN) * NHEAD), 256, 0, stream>>>(src, dst, alS, alD, mMax, E, NN);
  fill_f32_kernel<<<blk((long long)NN * NHEAD), 256, 0, stream>>>(denom, 0.0f, NN * NHEAD);
  gat_denom_kernel<<<blk((long long)(E + NN) * NHEAD), 256, 0, stream>>>(src, dst, alS, alD, mMax, denom, E, NN);
  fill_f32_kernel<<<blk((long long)NN * HDIM), 256, 0, stream>>>(bufA, 0.0f, NN * HDIM);
  gat_accum_kernel<<<blk((long long)(E + NN) * 32), 256, 0, stream>>>(src, dst, alS, alD, mMax, denom, hGat, bufA, E, NN);
  gat_fin_kernel<<<blk((long long)NN * HDIM), 256, 0, stream>>>(bufA, gatB, NN);

  // ---- pooling (mean + add share segment sum)
  fill_f32_kernel<<<blk((long long)G * HDIM), 256, 0, stream>>>(sums, 0.0f, G * HDIM);
  fill_f32_kernel<<<blk(G), 256, 0, stream>>>(cnts, 0.0f, G);
  pool_kernel<<<blk((long long)NN * 32), 256, 0, stream>>>(bufA, batch, sums, cnts, NN);

  // ---- MLP head + log_softmax (tiny: G=50 rows)
  head_kernel<<<G, HDIM, 0, stream>>>(sums, cnts, fc1W, fc1b, fc2W, fc2b, out);
}
